// deform_aggre_15668040696018
// MI455X (gfx1250) — matrix operand order repaired and checked by reading, not compiled, not run on hardware
//
#include <hip/hip_runtime.h>
#include <hip/hip_bf16.h>
#include <math.h>

typedef __attribute__((ext_vector_type(16))) _Float16 v16h;
typedef __attribute__((ext_vector_type(8)))  _Float16 v8h;
typedef __attribute__((ext_vector_type(8)))  float    v8f;

#define BB 8
#define CC 128
#define HH 96
#define WW 96
#define KK 9
#define HW (HH*WW)            // 9216
#define NPIX (BB*HW)          // 73728
#define KRED (CC*KK)          // 1152
#define NSTEP (KRED/32)       // 36
#define NTILE (CC/16)         // 8

// ---------------------------------------------------------------------------
// Prep: pack dcn weights into per-WMMA-fragment f16 layout.
// Bfrag[((s*8 + n)*32 + lane)*16 + j] = w[oc = n*16 + (lane&15)][i][k]
// where kk = (lane>>4)*16 + j, r = s*32 + kk, k = r>>7 (tap), i = r&127 (chan)
// ---------------------------------------------------------------------------
__global__ void prep_bfrag_kernel(const float* __restrict__ w, _Float16* __restrict__ bf) {
  int e = blockIdx.x * blockDim.x + threadIdx.x;     // 147456 total
  if (e >= NSTEP * NTILE * 32 * 16) return;
  int j    = e & 15;
  int lane = (e >> 4) & 31;
  int n    = (e >> 9) & 7;
  int s    = e >> 12;
  int r  = s * 32 + ((lane >> 4) * 16 + j);
  int k  = r >> 7;          // tap 0..8
  int i  = r & 127;         // input channel
  int oc = n * 16 + (lane & 15);
  bf[e] = (_Float16)w[(oc * CC + i) * KK + k];
}

// Per-channel fused BN params:  y = acc*S + T ; S=g/sqrt(v+eps), T=(b_conv-m)*S+beta
__global__ void prep_bn_kernel(const float* g, const float* beta, const float* m,
                               const float* v, const float* cb,
                               float* S, float* T) {
  int c = threadIdx.x;
  if (c < CC) {
    float s = g[c] * rsqrtf(v[c] + 1e-5f);
    S[c] = s;
    T[c] = (cb[c] - m[c]) * s + beta[c];
  }
}

// ---------------------------------------------------------------------------
// Offset conv: 3x3, pad 1, 27 outputs -> dy[b][k][hw], dx[b][k][hw], mask
// One thread per pixel; weight slice for each (kh,kw) staged in LDS (13.5KB).
// ---------------------------------------------------------------------------
__global__ void offsets_kernel(const float* __restrict__ in,
                               const float* __restrict__ ow,
                               const float* __restrict__ ob,
                               float* __restrict__ dyA, float* __restrict__ dxA,
                               float* __restrict__ mkA) {
  __shared__ float wsm[27 * CC];
  int p = blockIdx.x * blockDim.x + threadIdx.x;     // 73728
  int b   = p / HW;
  int rem = p - b * HW;
  int h   = rem / WW;
  int w   = rem - h * WW;

  float acc[27];
#pragma unroll
  for (int oc = 0; oc < 27; ++oc) acc[oc] = ob[oc];

  for (int kh = 0; kh < 3; ++kh) {
    for (int kw = 0; kw < 3; ++kw) {
      __syncthreads();
      for (int t = threadIdx.x; t < 27 * CC; t += blockDim.x)
        wsm[t] = ow[t * 9 + kh * 3 + kw];
      __syncthreads();
      int y = h - 1 + kh, x = w - 1 + kw;
      if (y >= 0 && y < HH && x >= 0 && x < WW) {
        const float* xb = in + ((size_t)b * CC) * HW + y * WW + x;
        for (int c = 0; c < CC; ++c) {
          float xv = xb[(size_t)c * HW];
#pragma unroll
          for (int oc = 0; oc < 27; ++oc)
            acc[oc] += xv * wsm[oc * CC + c];
        }
      }
    }
  }
  // om split: dy[k]=om[2k], dx[k]=om[2k+1], mask[k]=sigmoid(om[18+k])
#pragma unroll
  for (int k = 0; k < KK; ++k) {
    size_t o = ((size_t)b * KK + k) * HW + rem;
    dyA[o] = acc[2 * k];
    dxA[o] = acc[2 * k + 1];
    mkA[o] = 1.0f / (1.0f + __expf(-acc[18 + k]));
  }
}

// ---------------------------------------------------------------------------
// Fused deformable-sample + WMMA GEMM + BN + ReLU.
// Block = 256 thr = 8 waves; each wave owns a private 16-pixel M tile and a
// private 1KB LDS slice -> NO block barriers (CDNA5 per-wave LDS ops are
// in-order; compiler fences keep program order).
// Loop: tap k (9 iters, sampling state computed once) x channel slab (4,
// unrolled) x N-tile (8 WMMAs, 1-deep B software pipeline).
// ---------------------------------------------------------------------------
__global__ void dcn_gemm_kernel(const float* __restrict__ in,
                                const float* __restrict__ dyA,
                                const float* __restrict__ dxA,
                                const float* __restrict__ mkA,
                                const _Float16* __restrict__ bfrag,
                                const float* __restrict__ bnS,
                                const float* __restrict__ bnT,
                                float* __restrict__ out) {
  __shared__ __align__(16) _Float16 ldsA[8][16][32];   // 8 KB, 1KB per wave

  const int wv   = threadIdx.x >> 5;
  const int lane = threadIdx.x & 31;
  const int row  = lane & 15;         // pixel within wave tile / fragment row
  const int half = lane >> 4;

  const int pixBase = blockIdx.x * 128 + wv * 16;
  const int myPix   = pixBase + row;
  const int b   = myPix / HW;
  const int rem = myPix - b * HW;
  const int h   = rem / WW;
  const int w   = rem - h * WW;

  v8f acc[NTILE] = {};

  for (int k = 0; k < KK; ++k) {
    // ---- per-tap bilinear state (shared by the 4 channel slabs) ----
    size_t o = ((size_t)b * KK + k) * HW + rem;
    float dyv = dyA[o], dxv = dxA[o];
    float mval = mkA[o];
    float py = (float)(h - 1 + (k / 3)) + dyv;
    float px = (float)(w - 1 + (k % 3)) + dxv;
    float y0f = floorf(py), x0f = floorf(px);
    float fy = py - y0f, fx = px - x0f;
    int y0 = (int)y0f, x0 = (int)x0f;
    bool vy0 = (y0 >= 0) & (y0 < HH);
    bool vy1 = (y0 + 1 >= 0) & (y0 + 1 < HH);
    bool vx0 = (x0 >= 0) & (x0 < WW);
    bool vx1 = (x0 + 1 >= 0) & (x0 + 1 < WW);
    float w00 = (1.f - fy) * (1.f - fx) * (float)(vy0 && vx0);
    float w01 = (1.f - fy) * fx         * (float)(vy0 && vx1);
    float w10 = fy * (1.f - fx)         * (float)(vy1 && vx0);
    float w11 = fy * fx                 * (float)(vy1 && vx1);
    int yc0 = min(max(y0, 0), HH - 1), yc1 = min(max(y0 + 1, 0), HH - 1);
    int xc0 = min(max(x0, 0), WW - 1), xc1 = min(max(x0 + 1, 0), WW - 1);
    int i00 = yc0 * WW + xc0, i01 = yc0 * WW + xc1;
    int i10 = yc1 * WW + xc0, i11 = yc1 * WW + xc1;

#pragma unroll
    for (int cs = 0; cs < 4; ++cs) {
      const int s = k * 4 + cs;

      // ---- stage A tile: this lane covers 16 channels of its pixel ----
      {
        const int cbase = cs * 32 + half * 16;
        const float* xb = in + ((size_t)(b * CC + cbase)) * HW;
        _Float16* arow = &ldsA[wv][row][half * 16];
#pragma unroll
        for (int c2 = 0; c2 < 16; ++c2) {
          const float* xc = xb + (size_t)c2 * HW;
          float v = w00 * xc[i00] + w01 * xc[i01] + w10 * xc[i10] + w11 * xc[i11];
          arow[c2] = (_Float16)(v * mval);
        }
      }
      // wave-local ordering fence (slices are wave-private; per-wave LDS ops
      // are in-order on CDNA5, so no workgroup barrier is needed)
      asm volatile("" ::: "memory");

      // ---- A fragment, contiguous k (lanes 0-15: K0-15; 16-31: K16-31), matches bfrag
      const v8h* pr = (const v8h*)&ldsA[wv][row][0];
      v8h alo = pr[2 * half];
      v8h ahi = pr[2 * half + 1];
      v16h a = __builtin_shufflevector(alo, ahi, 0, 1, 2, 3, 4, 5, 6, 7,
                                       8, 9, 10, 11, 12, 13, 14, 15);

      // ---- prefetch next slab's B block while WMMAs run ----
      const v16h* bp = (const v16h*)bfrag + (size_t)s * (NTILE * 32);
      if (s + 1 < NSTEP)
        __builtin_prefetch((const char*)(bp + NTILE * 32) + lane * 256, 0, 0);

      // ---- 8 WMMAs with a 1-deep B-fragment software pipeline ----
      v16h bcur = bp[lane];
#pragma unroll
      for (int n = 0; n < NTILE; ++n) {
        v16h bnext = (n + 1 < NTILE) ? bp[(n + 1) * 32 + lane] : bcur;
        acc[n] = __builtin_amdgcn_wmma_f32_16x16x32_f16(
            false, a, false, bcur, (short)0, acc[n], false, false);
        bcur = bnext;
      }
      asm volatile("" ::: "memory");   // keep next-slab stores after A reads
    }
  }

  // ---- epilogue: BN + ReLU, C layout: VGPR v -> M row, lane -> N col ----
#pragma unroll
  for (int v = 0; v < 8; ++v) {
    int p  = pixBase + v + 8 * half;
    int pb = p / HW;
    int pr2 = p - pb * HW;
#pragma unroll
    for (int n = 0; n < NTILE; ++n) {
      int oc = n * 16 + row;
      float y = acc[n][v] * bnS[oc] + bnT[oc];
      y = fmaxf(y, 0.0f);
      out[((size_t)(pb * CC + oc)) * HW + pr2] = y;
    }
  }
}

// ---------------------------------------------------------------------------
// Global max / mean pooling over HW per (b,c)
// ---------------------------------------------------------------------------
__global__ void pool_kernel(const float* __restrict__ o2,
                            float* __restrict__ mx, float* __restrict__ av) {
  __shared__ float smax[256];
  __shared__ float ssum[256];
  int bc = blockIdx.x;
  const float* p = o2 + (size_t)bc * HW;
  float vmax = -1e30f, vsum = 0.f;
  for (int i = threadIdx.x; i < HW; i += blockDim.x) {
    float v = p[i];
    vmax = fmaxf(vmax, v);
    vsum += v;
  }
  smax[threadIdx.x] = vmax;
  ssum[threadIdx.x] = vsum;
  __syncthreads();
  for (int st = 128; st > 0; st >>= 1) {
    if (threadIdx.x < st) {
      smax[threadIdx.x] = fmaxf(smax[threadIdx.x], smax[threadIdx.x + st]);
      ssum[threadIdx.x] += ssum[threadIdx.x + st];
    }
    __syncthreads();
  }
  if (threadIdx.x == 0) {
    mx[bc] = smax[0];
    av[bc] = ssum[0] / (float)HW;
  }
}

// ---------------------------------------------------------------------------
// Channel-attention gate: sigmoid(relu(mx@W1^T)@W2^T + relu(av@W1^T)@W2^T)
// hidden = 32, one block.
// ---------------------------------------------------------------------------
__global__ void gate_kernel(const float* __restrict__ mx, const float* __restrict__ av,
                            const float* __restrict__ w1, const float* __restrict__ w2,
                            float* __restrict__ gate) {
  __shared__ float hs[BB][32];
  int t = threadIdx.x;               // 256 = 8 b * 32 j
  int b = t >> 5, j = t & 31;
  float sm = 0.f, sa = 0.f;
  for (int c = 0; c < CC; ++c) {
    float wv = w1[j * CC + c];
    sm += mx[b * CC + c] * wv;
    sa += av[b * CC + c] * wv;
  }
  hs[b][j] = fmaxf(sm, 0.f) + fmaxf(sa, 0.f);
  __syncthreads();
  for (int q = 0; q < 4; ++q) {
    int idx = t + 256 * q;           // 1024 = 8*128
    int gb = idx >> 7, gc = idx & 127;
    float s = 0.f;
#pragma unroll
    for (int jj = 0; jj < 32; ++jj) s += hs[gb][jj] * w2[gc * 32 + jj];
    gate[idx] = 1.0f / (1.0f + __expf(-s));
  }
}

__global__ void scale_kernel(float* __restrict__ out, const float* __restrict__ gate) {
  size_t i = (size_t)blockIdx.x * blockDim.x + threadIdx.x;
  if (i < (size_t)NPIX * CC) {
    int bc = (int)(i / HW);
    out[i] *= gate[bc];
  }
}

// ---------------------------------------------------------------------------
static inline size_t align256(size_t x) { return (x + 255) & ~(size_t)255; }

extern "C" void kernel_launch(void* const* d_in, const int* in_sizes, int n_in,
                              void* d_out, int out_size, void* d_ws, size_t ws_size,
                              hipStream_t stream) {
  const float* x      = (const float*)d_in[0];
  const float* off_w1 = (const float*)d_in[1];
  const float* off_b1 = (const float*)d_in[2];
  const float* dcn_w1 = (const float*)d_in[3];
  const float* dcn_b1 = (const float*)d_in[4];
  const float* bn_g1  = (const float*)d_in[5];
  const float* bn_b1  = (const float*)d_in[6];
  const float* bn_m1  = (const float*)d_in[7];
  const float* bn_v1  = (const float*)d_in[8];
  const float* off_w2 = (const float*)d_in[9];
  const float* off_b2 = (const float*)d_in[10];
  const float* dcn_w2 = (const float*)d_in[11];
  const float* dcn_b2 = (const float*)d_in[12];
  const float* bn_g2  = (const float*)d_in[13];
  const float* bn_b2  = (const float*)d_in[14];
  const float* bn_m2  = (const float*)d_in[15];
  const float* bn_v2  = (const float*)d_in[16];
  const float* cam_w1 = (const float*)d_in[17];
  const float* cam_w2 = (const float*)d_in[18];
  float* out = (float*)d_out;

  // workspace carve (~46.4 MB)
  char* ws = (char*)d_ws;
  size_t off = 0;
  float* o1 = (float*)(ws + off); off = align256(off + (size_t)NPIX * CC * 4);
  float* dy = (float*)(ws + off); off = align256(off + (size_t)BB * KK * HW * 4);
  float* dx = (float*)(ws + off); off = align256(off + (size_t)BB * KK * HW * 4);
  float* mk = (float*)(ws + off); off = align256(off + (size_t)BB * KK * HW * 4);
  _Float16* bf1 = (_Float16*)(ws + off); off = align256(off + (size_t)KRED * CC * 2);
  _Float16* bf2 = (_Float16*)(ws + off); off = align256(off + (size_t)KRED * CC * 2);
  float* S1 = (float*)(ws + off); off = align256(off + CC * 4);
  float* T1 = (float*)(ws + off); off = align256(off + CC * 4);
  float* S2 = (float*)(ws + off); off = align256(off + CC * 4);
  float* T2 = (float*)(ws + off); off = align256(off + CC * 4);
  float* mx = (float*)(ws + off); off = align256(off + BB * CC * 4);
  float* av = (float*)(ws + off); off = align256(off + BB * CC * 4);
  float* gate = (float*)(ws + off); off = align256(off + BB * CC * 4);
  (void)off; (void)ws_size; (void)in_sizes; (void)n_in; (void)out_size;

  // weight / BN prep
  prep_bfrag_kernel<<<(NSTEP * NTILE * 32 * 16) / 256, 256, 0, stream>>>(dcn_w1, bf1);
  prep_bfrag_kernel<<<(NSTEP * NTILE * 32 * 16) / 256, 256, 0, stream>>>(dcn_w2, bf2);
  prep_bn_kernel<<<1, 128, 0, stream>>>(bn_g1, bn_b1, bn_m1, bn_v1, dcn_b1, S1, T1);
  prep_bn_kernel<<<1, 128, 0, stream>>>(bn_g2, bn_b2, bn_m2, bn_v2, dcn_b2, S2, T2);

  // layer 1
  offsets_kernel<<<NPIX / 256, 256, 0, stream>>>(x, off_w1, off_b1, dy, dx, mk);
  dcn_gemm_kernel<<<NPIX / 128, 256, 0, stream>>>(x, dy, dx, mk, bf1, S1, T1, o1);

  // layer 2 (o2 written straight into d_out)
  offsets_kernel<<<NPIX / 256, 256, 0, stream>>>(o1, off_w2, off_b2, dy, dx, mk);
  dcn_gemm_kernel<<<NPIX / 128, 256, 0, stream>>>(o1, dy, dx, mk, bf2, S2, T2, out);

  // channel attention
  pool_kernel<<<BB * CC, 256, 0, stream>>>(out, mx, av);
  gate_kernel<<<1, 256, 0, stream>>>(mx, av, cam_w1, cam_w2, gate);
  scale_kernel<<<(NPIX * CC + 255) / 256, 256, 0, stream>>>(out, gate);
}
